// HiddenRepModel_52132313039428
// MI455X (gfx1250) — matrix operand order repaired and checked by reading, not compiled, not run on hardware
//
#include <hip/hip_runtime.h>
#include <hip/hip_bf16.h>
#include <math.h>

// ---------------- problem constants (match reference) ----------------
#define V_   100000
#define D_   256
#define S_   128
#define H_   512
#define B_   8192
#define K_   32
#define NR_  60000
#define NS_  40000   // V_ - NR_ : indices that use the MLP path

typedef __attribute__((ext_vector_type(16))) __bf16 v16bf;
typedef __attribute__((ext_vector_type(8)))  float  v8f;

static __device__ __forceinline__ __bf16 f2bf(float f) {
  return (__bf16)f;                                    // native v_cvt_*_bf16_f32
}
static __device__ __forceinline__ float bf2f(unsigned short s) {
  return __builtin_bit_cast(float, (unsigned)s << 16);
}
static __device__ __forceinline__ float softplus_f(float z) {
  return fmaxf(z, 0.f) + log1pf(expf(-fabsf(z)));      // -log_sigmoid(-z)
}
static __device__ __forceinline__ float wsum(float x) {  // wave32 all-reduce
#pragma unroll
  for (int o = 16; o > 0; o >>= 1) x += __shfl_xor(x, o, 32);
  return x;
}

// ---------------- kernel 1: pack weights into WMMA B-fragment layout --------
// B (16-bit, 32x16): fragment = 32 lanes x 16 halves. h = l>>4, n = l&15,
// half e -> k_local = 8h + e (e<8), 16 + 8h + (e-8) (e>=8). Fragment id = kb*NB + nb.
__global__ void pack_wmma_b(const float* __restrict__ src, __bf16* __restrict__ dst,
                            int Ktot, int Ntot) {
  int tid = blockIdx.x * blockDim.x + threadIdx.x;
  if (tid >= Ktot * Ntot) return;
  int frag = tid >> 9;            // 512 halves per fragment
  int w    = tid & 511;
  int l    = w >> 4;
  int e    = w & 15;
  int NB   = Ntot >> 4;
  int kb   = frag / NB;
  int nb   = frag - kb * NB;
  int k    = kb * 32 + ((l >= 16) ? 8 : 0) + ((e >> 3) << 4) + (e & 7);
  int n    = nb * 16 + (l & 15);
  dst[tid] = f2bf(src[k * Ntot + n]);
}

// ---------------- kernel 2: bf16-convert the 40000 special stoich rows ------
__global__ void conv_stoich(const float* __restrict__ st, __bf16* __restrict__ x) {
  int tid = blockIdx.x * blockDim.x + threadIdx.x;     // < NS_*S_
  x[tid] = f2bf(st[(size_t)NR_ * S_ + tid]);
}

// ---------------- kernel 3: fused 2-layer MLP for 40000 rows ----------------
// Block = 128 threads (4 waves). The block owns a 64-row M-tile (4 sub-tiles);
// each wave owns a disjoint N-slice, so every B fragment load feeds 4
// independent WMMAs (4x reuse + latency hiding). Hidden tile staged in LDS.
__global__ __launch_bounds__(128) void mlp40k(
    const __bf16* __restrict__ x,     // [NS_, S_] bf16 row-major
    const __bf16* __restrict__ w1p,   // packed fragments (S_/32)*(H_/16)
    const float*  __restrict__ b1,    // [H_]
    const __bf16* __restrict__ w2p,   // packed fragments (H_/32)*(D_/16)
    const float*  __restrict__ b2,    // [D_]
    __bf16* __restrict__ out)         // [NS_, D_] bf16 row-major
{
  __shared__ __align__(16) __bf16 hid[64 * H_];        // 64 KB, block-shared
  const int lane = threadIdx.x & 31;
  const int wave = threadIdx.x >> 5;
  const int hi   = lane >> 4;        // K-half-group selector for A frags
  const int rl   = lane & 15;        // A row / B,C column within tile
  const int m0   = blockIdx.x * 64;  // 40000 = 625 * 64, exact

  union Frag { v16bf v; uint4 u4[2]; };

  // per-M-subtile global row pointers for the A fragments of GEMM1
  const __bf16* xrow[4];
#pragma unroll
  for (int mt = 0; mt < 4; ++mt)
    xrow[mt] = x + (size_t)(m0 + mt * 16 + rl) * S_;

  // ---- GEMM1: hidden = relu(x @ w1 + b1), K = 128; wave owns 8 N-blocks ----
  const v16bf* w1f = (const v16bf*)w1p;
  for (int nb = wave * 8; nb < wave * 8 + 8; ++nb) {
    float bias = b1[nb * 16 + rl];
    v8f acc[4];
#pragma unroll
    for (int mt = 0; mt < 4; ++mt)
#pragma unroll
      for (int i = 0; i < 8; ++i) acc[mt][i] = bias;
#pragma unroll
    for (int kb = 0; kb < S_ / 32; ++kb) {
      v16bf bfr = w1f[(size_t)(kb * (H_ / 16) + nb) * 32 + lane];
#pragma unroll
      for (int mt = 0; mt < 4; ++mt) {
        Frag a;
        const __bf16* ap = xrow[mt] + kb * 32 + hi * 8;
        a.u4[0] = *(const uint4*)(ap);
        a.u4[1] = *(const uint4*)(ap + 16);
        acc[mt] = __builtin_amdgcn_wmma_f32_16x16x32_bf16(false, a.v, false, bfr,
                                                          (short)0, acc[mt],
                                                          false, false);
      }
    }
#pragma unroll
    for (int mt = 0; mt < 4; ++mt)
#pragma unroll
      for (int r = 0; r < 8; ++r)      // C layout: m = r + 8*hi, n = rl
        hid[(mt * 16 + r + hi * 8) * H_ + nb * 16 + rl] =
            f2bf(fmaxf(acc[mt][r], 0.f));
  }
  __syncthreads();

  // ---- GEMM2: out = hidden @ w2 + b2, K = 512; wave owns 4 N-blocks ----
  const v16bf* w2f = (const v16bf*)w2p;
  for (int nb = wave * 4; nb < wave * 4 + 4; ++nb) {
    float bias = b2[nb * 16 + rl];
    v8f acc[4];
#pragma unroll
    for (int mt = 0; mt < 4; ++mt)
#pragma unroll
      for (int i = 0; i < 8; ++i) acc[mt][i] = bias;
#pragma unroll
    for (int kb = 0; kb < H_ / 32; ++kb) {
      v16bf bfr = w2f[(size_t)(kb * (D_ / 16) + nb) * 32 + lane];
#pragma unroll
      for (int mt = 0; mt < 4; ++mt) {
        Frag a;
        const __bf16* ap = hid + (mt * 16 + rl) * H_ + kb * 32 + hi * 8;
        a.u4[0] = *(const uint4*)(ap);                 // ds_load_b128
        a.u4[1] = *(const uint4*)(ap + 16);
        acc[mt] = __builtin_amdgcn_wmma_f32_16x16x32_bf16(false, a.v, false, bfr,
                                                          (short)0, acc[mt],
                                                          false, false);
      }
    }
#pragma unroll
    for (int mt = 0; mt < 4; ++mt)
#pragma unroll
      for (int r = 0; r < 8; ++r)
        out[(size_t)(m0 + mt * 16 + r + hi * 8) * D_ + nb * 16 + rl] =
            f2bf(acc[mt][r]);
  }
}

// ---------------- kernel 4: zero the scalar output --------------------------
__global__ void zero1(float* o) { o[0] = 0.f; }

// ---------------- kernel 5: gather + masked skip-gram score -----------------
// One wave per pair; each lane owns 8 of the 256 embedding dims.
__global__ __launch_bounds__(256) void score_kernel(
    const int* __restrict__ pos_u, const int* __restrict__ pos_v,
    const int* __restrict__ neg_v,
    const float* __restrict__ u_emb, const float* __restrict__ v_emb,
    const __bf16* __restrict__ Mt,   const __bf16* __restrict__ Mc,
    float* __restrict__ outp)
{
  const int lane = threadIdx.x & 31;
  const int p    = blockIdx.x * 8 + (threadIdx.x >> 5);

  float u[8], v[8], nv[8];
  {
    int iu = pos_u[p];
    if (iu < NR_) {
      const float* r = u_emb + (size_t)iu * D_ + lane * 8;
      float4 a = *(const float4*)r, b = *(const float4*)(r + 4);
      u[0]=a.x; u[1]=a.y; u[2]=a.z; u[3]=a.w; u[4]=b.x; u[5]=b.y; u[6]=b.z; u[7]=b.w;
    } else {
      union { uint4 q; unsigned short s[8]; } t;
      t.q = *(const uint4*)(Mt + (size_t)(iu - NR_) * D_ + lane * 8);
#pragma unroll
      for (int i = 0; i < 8; ++i) u[i] = bf2f(t.s[i]);
    }
  }
  {
    int iv = pos_v[p];
    if (iv < NR_) {
      const float* r = v_emb + (size_t)iv * D_ + lane * 8;
      float4 a = *(const float4*)r, b = *(const float4*)(r + 4);
      v[0]=a.x; v[1]=a.y; v[2]=a.z; v[3]=a.w; v[4]=b.x; v[5]=b.y; v[6]=b.z; v[7]=b.w;
    } else {
      union { uint4 q; unsigned short s[8]; } t;
      t.q = *(const uint4*)(Mc + (size_t)(iv - NR_) * D_ + lane * 8);
#pragma unroll
      for (int i = 0; i < 8; ++i) v[i] = bf2f(t.s[i]);
    }
  }

  float d = 0.f;
#pragma unroll
  for (int i = 0; i < 8; ++i) d += u[i] * v[i];
  d = wsum(d);
  float zpos = fminf(10.f, fmaxf(-10.f, d));
  float acc = softplus_f(-zpos);                       // -log_sigmoid(pos)

  for (int k = 0; k < K_; ++k) {
    int in_ = neg_v[(size_t)p * K_ + k];
    if (in_ < NR_) {
      const float* r = v_emb + (size_t)in_ * D_ + lane * 8;
      float4 a = *(const float4*)r, b = *(const float4*)(r + 4);
      nv[0]=a.x; nv[1]=a.y; nv[2]=a.z; nv[3]=a.w; nv[4]=b.x; nv[5]=b.y; nv[6]=b.z; nv[7]=b.w;
    } else {
      union { uint4 q; unsigned short s[8]; } t;
      t.q = *(const uint4*)(Mc + (size_t)(in_ - NR_) * D_ + lane * 8);
#pragma unroll
      for (int i = 0; i < 8; ++i) nv[i] = bf2f(t.s[i]);
    }
    float dn = 0.f;
#pragma unroll
    for (int i = 0; i < 8; ++i) dn += u[i] * nv[i];
    dn = wsum(dn);
    float zn = fminf(10.f, fmaxf(-10.f, dn));
    acc += softplus_f(zn);                             // -log_sigmoid(-neg)
  }

  // 8-combo mask algebra collapses to a factor of 2; mean over B.
  if (lane == 0) atomicAdd(outp, acc * (2.0f / (float)B_));
}

// ---------------- launch -----------------------------------------------------
extern "C" void kernel_launch(void* const* d_in, const int* in_sizes, int n_in,
                              void* d_out, int out_size, void* d_ws, size_t ws_size,
                              hipStream_t stream) {
  (void)in_sizes; (void)n_in; (void)out_size; (void)ws_size;
  const int*   pos_u  = (const int*)d_in[0];
  const int*   pos_v  = (const int*)d_in[1];
  const int*   neg_v  = (const int*)d_in[2];
  const float* u_emb  = (const float*)d_in[3];
  const float* v_emb  = (const float*)d_in[4];
  const float* stoich = (const float*)d_in[5];
  const float* tw1 = (const float*)d_in[6];
  const float* tb1 = (const float*)d_in[7];
  const float* tw2 = (const float*)d_in[8];
  const float* tb2 = (const float*)d_in[9];
  const float* cw1 = (const float*)d_in[10];
  const float* cb1 = (const float*)d_in[11];
  const float* cw2 = (const float*)d_in[12];
  const float* cb2 = (const float*)d_in[13];

  char* ws = (char*)d_ws;
  size_t o = 0;
  __bf16* w1t = (__bf16*)(ws + o); o += (size_t)S_ * H_ * 2;          // 128 KB
  __bf16* w2t = (__bf16*)(ws + o); o += (size_t)H_ * D_ * 2;          // 256 KB
  __bf16* w1c = (__bf16*)(ws + o); o += (size_t)S_ * H_ * 2;
  __bf16* w2c = (__bf16*)(ws + o); o += (size_t)H_ * D_ * 2;
  __bf16* xbf = (__bf16*)(ws + o); o += (size_t)NS_ * S_ * 2;         // ~9.8 MB
  __bf16* Mt  = (__bf16*)(ws + o); o += (size_t)NS_ * D_ * 2;         // ~19.5 MB
  __bf16* Mc  = (__bf16*)(ws + o);                                     // ~19.5 MB

  pack_wmma_b<<<(S_ * H_) / 256, 256, 0, stream>>>(tw1, w1t, S_, H_);
  pack_wmma_b<<<(H_ * D_) / 256, 256, 0, stream>>>(tw2, w2t, H_, D_);
  pack_wmma_b<<<(S_ * H_) / 256, 256, 0, stream>>>(cw1, w1c, S_, H_);
  pack_wmma_b<<<(H_ * D_) / 256, 256, 0, stream>>>(cw2, w2c, H_, D_);
  conv_stoich<<<(NS_ * S_) / 256, 256, 0, stream>>>(stoich, xbf);

  mlp40k<<<NS_ / 64, 128, 0, stream>>>(xbf, w1t, tb1, w2t, tb2, Mt);
  mlp40k<<<NS_ / 64, 128, 0, stream>>>(xbf, w1c, cb1, w2c, cb2, Mc);

  zero1<<<1, 1, 0, stream>>>((float*)d_out);
  score_kernel<<<B_ / 8, 256, 0, stream>>>(pos_u, pos_v, neg_v, u_emb, v_emb,
                                           Mt, Mc, (float*)d_out);
}
